// ResidualMamba_82626580840542
// MI455X (gfx1250) — compile-verified, matrix operand order checked by reading, not run on hardware
//
#include <hip/hip_runtime.h>
#include <hip/hip_bf16.h>
#include <math.h>

#define D_MODEL 1024
#define D_INNER 2048
#define DT_RANK 64
#define D_STATE 16
#define BSZ     2
#define SEQ     1024
#define XDBL_W  (DT_RANK + 2 * D_STATE)   // 96

// CDNA5 async global->LDS copy (ASYNCcnt) if the toolchain exposes it.
#if __has_builtin(__builtin_amdgcn_global_load_async_to_lds_b128) && \
    __has_builtin(__builtin_amdgcn_s_wait_asynccnt)
#define USE_ASYNC_LDS 1
#else
#define USE_ASYNC_LDS 0
#endif

typedef __attribute__((ext_vector_type(16))) __bf16 v16bf;
typedef __attribute__((ext_vector_type(8)))  float  v8f;
typedef __attribute__((ext_vector_type(4)))  int    v4i;
typedef __attribute__((address_space(1))) v4i* gptr4;   // global int4*
typedef __attribute__((address_space(3))) v4i* lptr4;   // LDS int4*

// B-tile row stride in floats: 16B aligned (async b128 writes) and rotates
// LDS banks by 4 dwords per row to avoid 64-bank conflicts on fragment reads.
#define BROW_STRIDE 36

// ---------------------------------------------------------------------------
// 16x32 bf16 fragment from a row-major f32 matrix (ISA 7.12.2 16-bit A 16x32):
//   lane l -> row m=l&15; v16bf[0..7]=K=kbase+h8+{0..7}, [8..15]=+16
// ---------------------------------------------------------------------------
__device__ __forceinline__ v16bf ld_frag_bf16(const float* __restrict__ row, int kbase) {
  const int lane = threadIdx.x & 31;
  const int h8   = (lane & 16) >> 1;         // 0 or 8
  const float* q = row + kbase + h8;
  v16bf r;
#pragma unroll
  for (int j = 0; j < 8; ++j) r[j] = (__bf16)q[j];
#pragma unroll
  for (int j = 0; j < 8; ++j) r[j + 8] = (__bf16)q[j + 16];
  return r;
}

// Same fragment pattern, but sourced from an LDS-staged f32 tile (ds_load + cvt)
__device__ __forceinline__ v16bf ld_frag_lds(const float* __restrict__ tile, int t) {
  const int lane = threadIdx.x & 31;
  const float* q = tile + (t * 16 + (lane & 15)) * BROW_STRIDE + ((lane & 16) >> 1);
  v16bf r;
#pragma unroll
  for (int j = 0; j < 8; ++j) r[j] = (__bf16)q[j];
#pragma unroll
  for (int j = 0; j < 8; ++j) r[j + 8] = (__bf16)q[j + 16];
  return r;
}

__device__ __forceinline__ void stage_commit() {
#if USE_ASYNC_LDS
  __builtin_amdgcn_s_wait_asynccnt(0);       // s_wait_asynccnt 0
#endif
  __syncthreads();
}

// ---------------------------------------------------------------------------
// Block-cooperative WMMA GEMM: Out[M,N](+epi) = A[M,K] * W[N,K]^T
// 8 waves cover a 128Mx64N macro-tile; wave w owns rows [w*16, w*16+16).
// Per 32-K step the 64x32 f32 B tile is staged into double-buffered LDS with
// GLOBAL_LOAD_ASYNC_TO_LDS_B128 (fallback: load+ds_store), shared by all 8
// waves (8x less B traffic). A fragments load direct from global (L2-resident).
// MODE 0: store  MODE 1: softplus(acc+bias[n])  MODE 2: acc+resid
// Grid: (M/128) * (N/64) blocks of 256 threads.
// ---------------------------------------------------------------------------
template <int MODE>
__global__ __launch_bounds__(256) void wmma_gemm_lds(
    const float* __restrict__ A, int lda,
    const float* __restrict__ W, int ldw,
    float* __restrict__ Out, int ldo,
    const float* __restrict__ bias,
    const float* __restrict__ resid,
    int N, int K)
{
  __shared__ float bstage[2][64 * BROW_STRIDE];

  const int lane  = threadIdx.x & 31;
  const int wv    = threadIdx.x >> 5;                    // 0..7
  const int nblks = N >> 6;
  const int m0    = (blockIdx.x / nblks) * 128 + wv * 16;
  const int n0    = (blockIdx.x % nblks) * 64;

  const float* arow = A + (size_t)(m0 + (lane & 15)) * lda;

  // Stage 64 rows x 32 cols of W (f32) for K-block kb into bstage[buf].
  // 512 x 16B transfers; thread tid issues 2 contiguous-by-row copies.
  auto stage = [&](int kb, int buf) {
#pragma unroll
    for (int i = 0; i < 2; ++i) {
      const int g   = threadIdx.x * 2 + i;               // 0..511
      const int row = g >> 3;
      const int cg  = (g & 7) * 4;                       // float offset, 16B granule
      const float* src = W + (size_t)(n0 + row) * ldw + kb + cg;
      float* dst = &bstage[buf][row * BROW_STRIDE + cg];
#if USE_ASYNC_LDS
      float* s = const_cast<float*>(src);
      __builtin_amdgcn_global_load_async_to_lds_b128(
          (gptr4)s, (lptr4)dst, 0, 0);
#else
      *(float4*)dst = *(const float4*)src;
#endif
    }
  };

  v8f acc[4];
#pragma unroll
  for (int t = 0; t < 4; ++t)
    acc[t] = (v8f){0.f, 0.f, 0.f, 0.f, 0.f, 0.f, 0.f, 0.f};

  stage(0, 0);
  stage_commit();

  int ib = 0;
  for (int kb = 0; kb < K; kb += 32) {
    if (kb + 32 < K) stage(kb + 32, ib ^ 1);             // prefetch next K-block
    const v16bf a = ld_frag_bf16(arow, kb);
#pragma unroll
    for (int t = 0; t < 4; ++t) {
      const v16bf b = ld_frag_lds(bstage[ib], t);
      acc[t] = __builtin_amdgcn_wmma_f32_16x16x32_bf16(
          false, a, false, b, (short)0, acc[t], false, false);
    }
    stage_commit();                                      // wait asynccnt + barrier
    ib ^= 1;
  }

  // C/D layout: lane l -> col n=l&15; VGPR v -> row v + 8*(l>=16)
  const int nlo = lane & 15;
  const int mo  = (lane & 16) >> 1;
#pragma unroll
  for (int t = 0; t < 4; ++t) {
    const int ncol = n0 + t * 16 + nlo;
#pragma unroll
    for (int v = 0; v < 8; ++v) {
      const size_t oidx = (size_t)(m0 + mo + v) * ldo + ncol;
      float val = acc[t][v];
      if (MODE == 1) {
        val += bias[ncol];
        val = (val > 20.f) ? val : log1pf(__expf(val));
      } else if (MODE == 2) {
        val += resid[oidx];
      }
      Out[oidx] = val;
    }
  }
}

// ---------------------------------------------------------------------------
// Direct (no-LDS) WMMA GEMM for the skinny x_proj (N=96): one wave = 16x32.
// ---------------------------------------------------------------------------
template <int NT>
__global__ __launch_bounds__(256) void wmma_gemm_direct(
    const float* __restrict__ A, int lda,
    const float* __restrict__ W, int ldw,
    float* __restrict__ Out, int ldo,
    int N, int K)
{
  const int lane  = threadIdx.x & 31;
  const int wid   = blockIdx.x * (blockDim.x >> 5) + (threadIdx.x >> 5);
  const int ncols = N / (16 * NT);
  const int m0    = (wid / ncols) * 16;
  const int n0    = (wid % ncols) * 16 * NT;

  const float* arow = A + (size_t)(m0 + (lane & 15)) * lda;
  const float* brow[NT];
#pragma unroll
  for (int t = 0; t < NT; ++t)
    brow[t] = W + (size_t)(n0 + t * 16 + (lane & 15)) * ldw;

  v8f acc[NT];
#pragma unroll
  for (int t = 0; t < NT; ++t)
    acc[t] = (v8f){0.f, 0.f, 0.f, 0.f, 0.f, 0.f, 0.f, 0.f};

  for (int kb = 0; kb < K; kb += 32) {
    const v16bf a = ld_frag_bf16(arow, kb);
#pragma unroll
    for (int t = 0; t < NT; ++t) {
      const v16bf b = ld_frag_bf16(brow[t], kb);
      acc[t] = __builtin_amdgcn_wmma_f32_16x16x32_bf16(
          false, a, false, b, (short)0, acc[t], false, false);
    }
  }

  const int nlo = lane & 15;
  const int mo  = (lane & 16) >> 1;
#pragma unroll
  for (int t = 0; t < NT; ++t) {
    const int ncol = n0 + t * 16 + nlo;
#pragma unroll
    for (int v = 0; v < 8; ++v)
      Out[(size_t)(m0 + mo + v) * ldo + ncol] = acc[t][v];
  }
}

// ---------------------------------------------------------------------------
// Causal depthwise conv1d (D_CONV=4) + SiLU on the u-half of xz.
// ---------------------------------------------------------------------------
__global__ __launch_bounds__(256) void conv_silu_kernel(
    const float* __restrict__ xz, const float* __restrict__ cw,
    const float* __restrict__ cb, float* __restrict__ u)
{
  const int idx = blockIdx.x * blockDim.x + threadIdx.x;
  const int d = idx & (D_INNER - 1);
  const int l = (idx >> 11) & (SEQ - 1);
  const int b = idx >> 21;
  const float* col = xz + (size_t)b * SEQ * (2 * D_INNER) + d;
  float acc = cb[d];
#pragma unroll
  for (int j = 0; j < 4; ++j) {
    const int t = l - 3 + j;
    if (t >= 0) acc += col[(size_t)t * (2 * D_INNER)] * cw[d * 4 + j];
  }
  const float sig = 1.f / (1.f + __expf(-acc));
  u[idx] = acc * sig;
}

// ---------------------------------------------------------------------------
// Selective scan: one lane per (b, d) channel, 16-state in VGPRs, sequential
// over L, parallel over 4096 channels; fuses D-skip + SiLU(z) gate; writes
// in place over delta.
// ---------------------------------------------------------------------------
__global__ __launch_bounds__(256) void scan_kernel(
    const float* __restrict__ delta, const float* __restrict__ u,
    const float* __restrict__ xz, const float* __restrict__ xdbl,
    const float* __restrict__ A_log, const float* __restrict__ Dvec,
    float* __restrict__ yg)
{
  const int t = blockIdx.x * blockDim.x + threadIdx.x;
  const int b = t >> 11;
  const int d = t & (D_INNER - 1);

  float Arow[D_STATE], h[D_STATE];
#pragma unroll
  for (int n = 0; n < D_STATE; ++n) {
    Arow[n] = -__expf(A_log[d * D_STATE + n]);
    h[n] = 0.f;
  }
  const float Dd = Dvec[d];

  const float* drow = delta + (size_t)b * SEQ * D_INNER + d;
  const float* urow = u     + (size_t)b * SEQ * D_INNER + d;
  const float* zrow = xz    + (size_t)b * SEQ * (2 * D_INNER) + D_INNER + d;
  const float* bc   = xdbl  + (size_t)b * SEQ * XDBL_W + DT_RANK;
  float* yrow       = yg    + (size_t)b * SEQ * D_INNER + d;

  for (int l = 0; l < SEQ; ++l) {
    __builtin_prefetch(drow + (size_t)(l + 8) * D_INNER, 0, 0);
    __builtin_prefetch(urow + (size_t)(l + 8) * D_INNER, 0, 0);
    const float dl = drow[(size_t)l * D_INNER];
    const float ul = urow[(size_t)l * D_INNER];
    const float du = dl * ul;
    const float* Bl = bc + (size_t)l * XDBL_W;
    const float* Cl = Bl + D_STATE;
    float y = 0.f;
#pragma unroll
    for (int n = 0; n < D_STATE; ++n) {
      const float dA = __expf(dl * Arow[n]);
      h[n] = dA * h[n] + du * Bl[n];
      y = fmaf(h[n], Cl[n], y);
    }
    const float z   = zrow[(size_t)l * (2 * D_INNER)];
    const float sig = 1.f / (1.f + __expf(-z));
    yrow[(size_t)l * D_INNER] = (y + ul * Dd) * (z * sig);
  }
}

// ---------------------------------------------------------------------------
extern "C" void kernel_launch(void* const* d_in, const int* in_sizes, int n_in,
                              void* d_out, int out_size, void* d_ws, size_t ws_size,
                              hipStream_t stream) {
  const float* x          = (const float*)d_in[0];
  const float* in_proj_w  = (const float*)d_in[1];
  const float* conv_w     = (const float*)d_in[2];
  const float* conv_b     = (const float*)d_in[3];
  const float* x_proj_w   = (const float*)d_in[4];
  const float* dt_proj_w  = (const float*)d_in[5];
  const float* dt_proj_b  = (const float*)d_in[6];
  const float* A_log      = (const float*)d_in[7];
  const float* Dvec       = (const float*)d_in[8];
  const float* out_proj_w = (const float*)d_in[9];
  float* out = (float*)d_out;

  // Workspace (floats): xz | u | x_dbl | delta (reused in-place as gated y)
  float* xz    = (float*)d_ws;                             // 2*1024*4096
  float* u     = xz    + (size_t)BSZ * SEQ * 2 * D_INNER;  // 2*1024*2048
  float* xdbl  = u     + (size_t)BSZ * SEQ * D_INNER;      // 2*1024*96
  float* delta = xdbl  + (size_t)BSZ * SEQ * XDBL_W;       // 2*1024*2048

  // 1) in_proj: xz = x @ in_proj_w^T        M=2048 N=4096 K=1024
  //    grid = (2048/128)*(4096/64) = 16*64 = 1024
  wmma_gemm_lds<0><<<1024, 256, 0, stream>>>(
      x, D_MODEL, in_proj_w, D_MODEL, xz, 2 * D_INNER, nullptr, nullptr,
      2 * D_INNER, D_MODEL);

  // 2) causal depthwise conv + SiLU -> u
  conv_silu_kernel<<<(BSZ * SEQ * D_INNER) / 256, 256, 0, stream>>>(
      xz, conv_w, conv_b, u);

  // 3) x_proj: x_dbl = u @ x_proj_w^T       M=2048 N=96 K=2048 (direct, skinny)
  //    waves = 128*(96/32) = 384 -> 48 blocks
  wmma_gemm_direct<2><<<48, 256, 0, stream>>>(
      u, D_INNER, x_proj_w, D_INNER, xdbl, XDBL_W, XDBL_W, D_INNER);

  // 4) dt_proj: delta = softplus(x_dbl[:, :64] @ dt_proj_w^T + b)
  //    grid = 16*(2048/64) = 512
  wmma_gemm_lds<1><<<512, 256, 0, stream>>>(
      xdbl, XDBL_W, dt_proj_w, DT_RANK, delta, D_INNER, dt_proj_b, nullptr,
      D_INNER, DT_RANK);

  // 5) selective scan + D-skip + SiLU(z) gate (in-place over delta)
  scan_kernel<<<(BSZ * D_INNER) / 256, 256, 0, stream>>>(
      delta, u, xz, xdbl, A_log, Dvec, delta);

  // 6) out_proj + residual: out = x + yg @ out_proj_w^T  M=2048 N=1024 K=2048
  //    grid = 16*(1024/64) = 256
  wmma_gemm_lds<2><<<256, 256, 0, stream>>>(
      delta, D_INNER, out_proj_w, D_INNER, out, D_MODEL, nullptr, x,
      D_MODEL, D_INNER);

  (void)in_sizes; (void)n_in; (void)out_size; (void)ws_size;
}
